// SpatialConv3d_9929964388600
// MI455X (gfx1250) — matrix operand order repaired and checked by reading, not compiled, not run on hardware
//
#include <hip/hip_runtime.h>

// ---------------------------------------------------------------------------
// CDNA5 (gfx1250) directional scan-conv:  y[t] = x[t] + relu(conv3x3(y[t-1]))
// GEMM per step: M=64 oc  x  N=4096 pixels  x  K=576 (9 taps * 64 ic)
// v_wmma_f32_16x16x32_bf16, A = pre-packed weights, B = bf16 LDS tile.
// Tensor is L2-resident (67MB < 192MB); the scan is latency-bound, so the
// goal is minimum instructions between dependent steps: vector staging loads,
// packed A-fragments, back-to-back WMMAs.
// ---------------------------------------------------------------------------

typedef __bf16 bf16;
typedef __attribute__((ext_vector_type(16))) __bf16 v16bf;
typedef __attribute__((ext_vector_type(8)))  float   v8f;

#define SC   (64*64*64)          // channel stride in elements
#define NOCT 4                   // 64 oc / 16
#define NCHK 18                  // 576 K / 32
#define FRAG (NOCT*NCHK*32*16)   // packed bf16 elements per weight tensor

// ---------------------------------------------------------------------------
// Pack w[oc][ic][3][3] (fp32) into the dense 16x32 bf16 A-matrix lane layout
// (this code's contiguous k order, same as B): lane<16: M=lane, K = e ; lane>=16: K = 16+e.
// K ordering: k = (ky*3+kx)*64 + ic  -> chunk c covers K0=c*32.
// ---------------------------------------------------------------------------
__global__ void prep_wfrag_kernel(const float* __restrict__ w,
                                  bf16* __restrict__ dst) {
    int idx = blockIdx.x * 256 + threadIdx.x;
    if (idx >= FRAG) return;
    int e    = idx & 15;
    int lane = (idx >> 4) & 31;
    int rest = idx >> 9;            // 0..71
    int c    = rest % NCHK;         // K chunk
    int oct  = rest / NCHK;         // octile
    int hi   = lane >> 4;
    int oc   = oct * 16 + (lane & 15);
    int klocal = e + hi * 16;                        // 0..31
    int ic   = ((c & 1) << 5) + klocal;              // k % 64
    int kk   = c >> 1;                               // tap index 0..8
    int ky   = kk / 3, kx = kk % 3;
    dst[idx] = (bf16)w[(oc * 64 + ic) * 9 + ky * 3 + kx];
}

// ---------------------------------------------------------------------------
// One scan step on a 16x16 pixel tile: 8 waves, wave = (octile, 8 pixel rows).
//  prev: carry slice base (already updated),  cur: in/out slice (in-place).
//  VEC: plane fast axis is contiguous (s_v==1) -> float4 halo staging.
// ---------------------------------------------------------------------------
template <bool VEC>
__global__ __launch_bounds__(256) void scan_step_kernel(
    const float* __restrict__ prev,
    float*       __restrict__ cur,
    const bf16*  __restrict__ wfrag,
    int s_u, int s_v)
{
    __shared__ __align__(32) bf16 lds[18 * 18 * 64];   // halo tile, [u][v][ic]

    const int tid    = threadIdx.x;
    const int tile_u = (blockIdx.x >> 2) << 4;
    const int tile_v = (blockIdx.x & 3)  << 4;

    // ---- stage carry halo (18x18 pixels x 64ch) as bf16, zero padded ------
    if constexpr (VEC) {
        // Interior 16 columns: aligned float4 along v (s_v == 1).
        // 18(lu) * 64(ic) * 4(quad) = 4608 tasks, exactly 18 per thread.
        for (int i = tid; i < 18 * 64 * 4; i += 256) {
            int quad = i & 3;
            int rest = i >> 2;
            int ic   = rest & 63;
            int lu   = rest >> 6;                 // 0..17
            int gu   = tile_u + lu - 1;
            float4 val = make_float4(0.f, 0.f, 0.f, 0.f);
            if ((unsigned)gu < 64u)
                val = *(const float4*)&prev[(size_t)ic * SC + (size_t)gu * s_u
                                            + (tile_v + quad * 4)];
            int base = (lu * 18 + 1 + quad * 4) * 64 + ic;
            lds[base      ] = (bf16)val.x;
            lds[base + 64 ] = (bf16)val.y;
            lds[base + 128] = (bf16)val.z;
            lds[base + 192] = (bf16)val.w;
        }
        // Edge columns lv==0 (gv=tile_v-1) and lv==17 (gv=tile_v+16):
        // 18 * 64 * 2 = 2304 tasks, exactly 9 per thread.
        for (int i = tid; i < 18 * 64 * 2; i += 256) {
            int side = i & 1;
            int rest = i >> 1;
            int ic   = rest & 63;
            int lu   = rest >> 6;
            int gu   = tile_u + lu - 1;
            int gv   = side ? (tile_v + 16) : (tile_v - 1);
            float val = 0.f;
            if ((unsigned)gu < 64u && (unsigned)gv < 64u)
                val = prev[(size_t)ic * SC + (size_t)gu * s_u + gv];
            lds[(lu * 18 + (side ? 17 : 0)) * 64 + ic] = (bf16)val;
        }
    } else {
        for (int i = tid; i < 18 * 18 * 64; i += 256) {
            int lv = i % 18;
            int r2 = i / 18;
            int lu = r2 % 18;
            int ic = r2 / 18;
            int gu = tile_u + lu - 1;
            int gv = tile_v + lv - 1;
            float val = 0.0f;
            if ((unsigned)gu < 64u && (unsigned)gv < 64u)
                val = prev[(size_t)ic * SC + (size_t)gu * s_u + (size_t)gv * s_v];
            lds[(lu * 18 + lv) * 64 + ic] = (bf16)val;
        }
    }
    __syncthreads();

    const int wave = tid >> 5;
    const int lane = tid & 31;
    const int oct  = wave & 3;              // which 16 output channels
    const int r0   = (wave >> 2) << 3;      // pixel rows 0..7 or 8..15
    const int vl   = lane & 15;             // D-matrix column -> pixel v
    const int hi   = lane >> 4;

    v8f acc[8] = {};                        // 8 row-tiles of 16x16 f32

    // A fragments for this octile, one v16bf (32B) per chunk per lane
    const v16bf* wf = (const v16bf*)wfrag + (size_t)oct * NCHK * 32 + lane;

    #pragma unroll
    for (int c = 0; c < NCHK; ++c) {
        v16bf a = wf[c * 32];               // weights: reused across 8 rows
        int kk = c >> 1;
        int ky = (kk >= 6) ? 2 : ((kk >= 3) ? 1 : 0);
        int kx = kk - ky * 3;
        // B lane layout: lane<16 -> K = K0+e, lane>=16 -> K = K0+16+e
        // => 16 contiguous channels starting at:
        int ic0 = ((c & 1) << 5) + (hi << 4);

        #pragma unroll
        for (int r = 0; r < 8; ++r) {
            const bf16* p = &lds[((r0 + r + ky) * 18 + (vl + kx)) * 64 + ic0];
            v16bf b = *(const v16bf*)p;     // 32B aligned LDS read
            acc[r] = __builtin_amdgcn_wmma_f32_16x16x32_bf16(
                         false, a, false, b, (short)0, acc[r], false, false);
        }
    }

    // ---- epilogue: cur = cur + relu(acc), coalesced over pixels -----------
    #pragma unroll
    for (int r = 0; r < 8; ++r) {
        int u = tile_u + r0 + r;
        #pragma unroll
        for (int q = 0; q < 8; ++q) {
            int oc = oct * 16 + q + (hi << 3);     // D: M = q (+8 for hi half)
            int v  = tile_v + vl;                  // D: N = lane&15
            size_t off = (size_t)oc * SC + (size_t)u * s_u + (size_t)v * s_v;
            float z = acc[r][q];
            cur[off] = cur[off] + (z > 0.0f ? z : 0.0f);
        }
    }
}

// ---------------------------------------------------------------------------
extern "C" void kernel_launch(void* const* d_in, const int* in_sizes, int n_in,
                              void* d_out, int out_size, void* d_ws, size_t ws_size,
                              hipStream_t stream)
{
    const float* x = (const float*)d_in[0];
    float* out = (float*)d_out;
    const size_t total = (size_t)64 * 64 * 64 * 64;

    // y starts as a copy of x; each pass updates it in place slice by slice.
    hipMemcpyAsync(out, x, total * sizeof(float), hipMemcpyDeviceToDevice, stream);

    // Pack all six weight tensors into WMMA A-fragment layout (bf16) in d_ws.
    bf16* wf = (bf16*)d_ws;
    for (int p = 0; p < 6; ++p) {
        prep_wfrag_kernel<<<(FRAG + 255) / 256, 256, 0, stream>>>(
            (const float*)d_in[1 + p], wf + (size_t)p * FRAG);
    }

    // pass order: H fwd, H rev, W fwd, W rev, D fwd, D rev
    struct Pass { int s_t, s_u, s_v, rev; };
    const Pass passes[6] = {
        {   64, 4096,  1, 0 },   // w_ud: scan H, plane (D,W)
        {   64, 4096,  1, 1 },   // w_du
        {    1, 4096, 64, 0 },   // w_lr: scan W, plane (D,H)
        {    1, 4096, 64, 1 },   // w_rl
        { 4096,   64,  1, 0 },   // w_fb: scan D, plane (H,W)
        { 4096,   64,  1, 1 },   // w_bf
    };

    for (int p = 0; p < 6; ++p) {
        const Pass& P = passes[p];
        for (int s = 1; s < 64; ++s) {
            int t  = P.rev ? (63 - s) : s;
            int tp = P.rev ? (t + 1) : (t - 1);
            const float* prev = out + (size_t)tp * P.s_t;
            float*       cur  = out + (size_t)t  * P.s_t;
            const bf16*  w    = wf + (size_t)p * FRAG;
            if (P.s_v == 1)
                scan_step_kernel<true ><<<16, 256, 0, stream>>>(prev, cur, w, P.s_u, P.s_v);
            else
                scan_step_kernel<false><<<16, 256, 0, stream>>>(prev, cur, w, P.s_u, P.s_v);
        }
    }
}
